// SSMBlock3_2d_73194832658532
// MI455X (gfx1250) — compile-verified, matrix operand order checked by reading, not run on hardware
//
#include <hip/hip_runtime.h>
#include <hip/hip_bf16.h>
#include <math.h>

typedef __attribute__((ext_vector_type(16))) _Float16 v16h;
typedef __attribute__((ext_vector_type(8)))  _Float16 v8h;
typedef __attribute__((ext_vector_type(8)))  float    v8f;

#define HH 32
#define WW 32
#define CC 128
#define GG 4
#define GC 32
#define KK 15
#define K2 225
#define NPIX 1024
#define NOFF_PAD 1808   // 1800 padded to 16
#define NMSK_PAD 912    // 900 padded to 16

// WMMA 16-bit A/B fragment K-offset map (ISA 7.12.2): lane half selects K group
__device__ __forceinline__ int wmma_kmap(int i, int half) {
  int v = i >> 1, r = i & 1;
  int k = (v < 4) ? (2 * v + r) : (16 + 2 * (v - 4) + r);
  return k + half * 8;
}

// ---------------------------------------------------------------------------
// Small precompute: sigma -> env (normalized Gaussian), kernel MLP -> kw
// ---------------------------------------------------------------------------
__global__ void prep_kernel(const float* __restrict__ raw_sigma,
                            const float* __restrict__ k1w, const float* __restrict__ k1b,
                            const float* __restrict__ k2w, const float* __restrict__ k2b,
                            const float* __restrict__ k3w, const float* __restrict__ k3b,
                            float* __restrict__ env, float* __restrict__ kwout) {
  __shared__ float se[K2];
  __shared__ float stot;
  int t = threadIdx.x;
  float sigma = log1pf(expf(raw_sigma[0]));
  sigma = fminf(fmaxf(sigma, 0.001f), 0.5f);
  if (t < K2) {
    int kh = t / KK, kw = t % KK;
    float gh = -0.5f + kh * (1.0f / 14.0f);
    float gw = -0.5f + kw * (1.0f / 14.0f);
    se[t] = expf(-(gh * gh + gw * gw) / (2.0f * sigma * sigma));
  }
  __syncthreads();
  if (t == 0) {
    float s = 0.0f;
    for (int i = 0; i < K2; ++i) s += se[i];
    stot = fmaxf(s, 1e-8f);
  }
  __syncthreads();
  if (t < K2) {
    env[t] = se[t] / stot;
    int kh = t / KK, kw = t % KK;
    float p0 = (-0.5f + kh * (1.0f / 14.0f)) * 2.0f;
    float p1 = (-0.5f + kw * (1.0f / 14.0f)) * 2.0f;
    float h1[32], h2[32];
    for (int j = 0; j < 32; ++j) {
      float v = p0 * k1w[j] + p1 * k1w[32 + j] + k1b[j];
      h1[j] = v / (1.0f + expf(-v));
    }
    for (int j = 0; j < 32; ++j) {
      float v = k2b[j];
      for (int i = 0; i < 32; ++i) v += h1[i] * k2w[i * 32 + j];
      h2[j] = v / (1.0f + expf(-v));
    }
    for (int j = 0; j < 128; ++j) {
      float v = k3b[j];
      for (int i = 0; i < 32; ++i) v += h2[i] * k3w[i * 128 + j];
      kwout[t * 128 + j] = v;  // flat (K,K,G,GC) order == (G,K2,GC) flat reinterpret
    }
  }
}

// ---------------------------------------------------------------------------
// Pack B (f32 [128][N] row-major) into WMMA-fragment order, f16, zero-padded:
//   dst[(((ntile*4 + kstep)*32 + lane)*16 + i] = B[kmap(i,lane>>4)+32*kstep][16*ntile + (lane&15)]
// Each GEMM lane then reads its whole fragment as one contiguous 32B chunk.
// ---------------------------------------------------------------------------
__global__ void pack_b_f16(const float* __restrict__ src, _Float16* __restrict__ dst,
                           int N, int Npad) {
  int id = blockIdx.x * blockDim.x + threadIdx.x;   // one thread per (ntile,kstep,lane)
  int Ntiles = Npad >> 4;
  if (id >= Ntiles * 4 * 32) return;
  int t = id >> 7;
  int rem = id & 127;
  int s = rem >> 5;
  int l = rem & 31;
  int half = l >> 4;
  int n = (t << 4) + (l & 15);
  _Float16* d = dst + (size_t)id * 16;
#pragma unroll
  for (int i = 0; i < 16; ++i) {
    int k = wmma_kmap(i, half) + 32 * s;
    d[i] = (n < N) ? (_Float16)src[k * N + n] : (_Float16)0.0f;
  }
}

__global__ void pad_bias_f32(const float* __restrict__ src, float* __restrict__ dst,
                             int N, int Npad) {
  int id = blockIdx.x * blockDim.x + threadIdx.x;
  if (id >= Npad) return;
  dst[id] = (id < N) ? src[id] : 0.0f;
}

// ---------------------------------------------------------------------------
// Depthwise 3x3 conv + SiLU -> f16, and x -> f16
// ---------------------------------------------------------------------------
__global__ void dwconv_silu(const float* __restrict__ x, const float* __restrict__ dwk,
                            const float* __restrict__ dwb,
                            _Float16* __restrict__ x_h, _Float16* __restrict__ y_h) {
  int id = blockIdx.x * blockDim.x + threadIdx.x;
  if (id >= NPIX * CC) return;
  int c = id & (CC - 1);
  int pix = id >> 7;
  int w = pix & (WW - 1), h = pix >> 5;
  float acc = dwb[c];
#pragma unroll
  for (int kh = 0; kh < 3; ++kh)
#pragma unroll
    for (int kw = 0; kw < 3; ++kw) {
      int hh = h + kh - 1, wp = w + kw - 1;
      if (hh < 0 || hh >= HH || wp < 0 || wp >= WW) continue;
      acc += x[(((hh << 5) + wp) << 7) + c] * dwk[(kh * 3 + kw) * CC + c];
    }
  float s = acc / (1.0f + expf(-acc));
  y_h[id] = (_Float16)s;
  x_h[id] = (_Float16)x[id];
}

// ---------------------------------------------------------------------------
// WMMA GEMM: out[M,Npad] = A[M,128](f16,row-major) @ Bpack(fragment order) +bias,*scale
// One wave per 16x16 tile; K=128 -> 4 x v_wmma_f32_16x16x32_f16.
// A: two contiguous 16B chunks per lane; B: one contiguous 32B chunk per lane.
// ---------------------------------------------------------------------------
__global__ void wmma_gemm(const _Float16* __restrict__ A, const _Float16* __restrict__ Bp,
                          const float* __restrict__ bias, const float* __restrict__ scale_p,
                          float* __restrict__ outF, _Float16* __restrict__ outH,
                          int Npad) {
  const int K = 128;
  int wave = threadIdx.x >> 5;
  int lane = threadIdx.x & 31;
  int Ntiles = Npad >> 4;
  int ntile = blockIdx.x * 4 + wave;
  if (ntile >= Ntiles) return;
  int mtile = blockIdx.y;
  int m0 = mtile << 4, n0 = ntile << 4;
  int half = lane >> 4, mn = lane & 15;

  const _Float16* Arow = A + (size_t)(m0 + mn) * K + half * 8;
  const v16h* Bfrag = (const v16h*)Bp + ((size_t)ntile * 4) * 32 + lane;

  v8f c = {};
#pragma unroll
  for (int ks = 0; ks < 4; ++ks) {
    v8h a_lo = *(const v8h*)(Arow + ks * 32);
    v8h a_hi = *(const v8h*)(Arow + ks * 32 + 16);
    v16h a = __builtin_shufflevector(a_lo, a_hi, 0, 1, 2, 3, 4, 5, 6, 7,
                                     8, 9, 10, 11, 12, 13, 14, 15);
    v16h b = Bfrag[ks * 32];
    c = __builtin_amdgcn_wmma_f32_16x16x32_f16(false, a, false, b, (short)0, c,
                                               false, false);
  }

  float scl = scale_p ? scale_p[0] : 1.0f;
  float bv = bias ? bias[n0 + mn] : 0.0f;
#pragma unroll
  for (int r = 0; r < 8; ++r) {
    int m = m0 + r + half * 8;                       // C/D layout: vgpr r -> row r / r+8
    float v = (c[r] + bv) * scl;
    size_t idx = (size_t)m * Npad + n0 + mn;
    if (outF) outF[idx] = v;
    if (outH) outH[idx] = (_Float16)v;
  }
}

// ---------------------------------------------------------------------------
// Softmax over 225 taps per (pixel, group); mask row scaled by env; in-place
// ---------------------------------------------------------------------------
__global__ void softmax_mask(float* __restrict__ mask, const float* __restrict__ env) {
  int gtid = blockIdx.x * blockDim.x + threadIdx.x;
  int wave = gtid >> 5;
  int lane = threadIdx.x & 31;
  int pixel = wave >> 2, g = wave & 3;
  float* row = mask + (size_t)pixel * NMSK_PAD + g * K2;

  float vals[8];
  float mx = -1e30f;
#pragma unroll
  for (int i = 0; i < 8; ++i) {
    int k = lane + i * 32;
    float v = (k < K2) ? row[k] * env[k] : -1e30f;
    vals[i] = v;
    mx = fmaxf(mx, v);
  }
  for (int s = 16; s > 0; s >>= 1) mx = fmaxf(mx, __shfl_xor(mx, s, 32));
  float sum = 0.0f;
#pragma unroll
  for (int i = 0; i < 8; ++i) {
    int k = lane + i * 32;
    float e = (k < K2) ? expf(vals[i] - mx) : 0.0f;
    vals[i] = e;
    sum += e;
  }
  for (int s = 16; s > 0; s >>= 1) sum += __shfl_xor(sum, s, 32);
  float inv = (sum > 0.0f && isfinite(sum)) ? 1.0f / sum : 0.0f;
#pragma unroll
  for (int i = 0; i < 8; ++i) {
    int k = lane + i * 32;
    if (k < K2) {
      float r = vals[i] * inv;
      if (!isfinite(r)) r = 0.0f;
      row[k] = r;
    }
  }
}

// ---------------------------------------------------------------------------
// Deformable bilinear gather + attention + learned kernel weights
// one wave per (pixel, group); lane = channel within group (GC==32==wave32)
// ---------------------------------------------------------------------------
__global__ void deform_sample(const float* __restrict__ x_proj,
                              const float* __restrict__ offs,
                              const float* __restrict__ attn,
                              const float* __restrict__ kw,
                              float* __restrict__ out_pre) {
  int gtid = blockIdx.x * blockDim.x + threadIdx.x;
  int wave = gtid >> 5;
  int lane = threadIdx.x & 31;
  int pixel = wave >> 2, g = wave & 3;
  int h = pixel >> 5, w = pixel & (WW - 1);
  const float* orow = offs + (size_t)pixel * NOFF_PAD + g * (K2 * 2);
  const float* arow = attn + (size_t)pixel * NMSK_PAD + g * K2;
  const float* krow = kw + g * K2 * GC;
  int baseC = g * GC + lane;
  float acc = 0.0f;
  for (int k = 0; k < K2; ++k) {
    float a = arow[k];
    float o0 = orow[2 * k];
    float o1 = orow[2 * k + 1];
    int kh = k / KK, kwi = k % KK;
    float abs_h = (float)h + (float)(kh - 7) + o0;
    float abs_w = (float)w + (float)(kwi - 7) + o1;
    float valid = (abs_h < 0.0f || abs_h > (float)(HH - 1) ||
                   abs_w < 0.0f || abs_w > (float)(WW - 1)) ? 0.0f : 1.0f;
    float ah = fminf(fmaxf(abs_h, 0.0f), (float)(HH - 1));
    float aw = fminf(fmaxf(abs_w, 0.0f), (float)(WW - 1));
    int hf = min((int)ah, HH - 1);
    int wf = min((int)aw, WW - 1);
    int hc = min(hf + 1, HH - 1);
    int wc = min(wf + 1, WW - 1);
    float hw = ah - (float)hf;
    float wwt = aw - (float)wf;
    float s00 = x_proj[((((hf << 5) + wf) << 7)) + baseC];
    float s01 = x_proj[((((hf << 5) + wc) << 7)) + baseC];
    float s10 = x_proj[((((hc << 5) + wf) << 7)) + baseC];
    float s11 = x_proj[((((hc << 5) + wc) << 7)) + baseC];
    float samp = (s00 * (1.0f - hw) * (1.0f - wwt) +
                  s01 * (1.0f - hw) * wwt +
                  s10 * hw * (1.0f - wwt) +
                  s11 * hw * wwt) * valid;
    acc += a * krow[k * GC + lane] * samp;
  }
  out_pre[(size_t)pixel * CC + baseC] = acc;
}

// ---------------------------------------------------------------------------
// Squeeze-Excite gate: global mean -> 32 (SiLU) -> 128 (sigmoid)
// ---------------------------------------------------------------------------
__global__ void se_kernel(const float* __restrict__ out_pre,
                          const float* __restrict__ se1w, const float* __restrict__ se1b,
                          const float* __restrict__ se2w, const float* __restrict__ se2b,
                          float* __restrict__ scale) {
  __shared__ float sp[CC];
  __shared__ float sh[32];
  int t = threadIdx.x;  // 128 threads
  float s = 0.0f;
  for (int p = 0; p < NPIX; ++p) s += out_pre[p * CC + t];
  sp[t] = s * (1.0f / (float)NPIX);
  __syncthreads();
  if (t < 32) {
    float v = se1b[t];
    for (int c = 0; c < CC; ++c) v += sp[c] * se1w[c * 32 + t];
    sh[t] = v / (1.0f + expf(-v));
  }
  __syncthreads();
  float v = se2b[t];
  for (int j = 0; j < 32; ++j) v += sh[j] * se2w[j * CC + t];
  scale[t] = 1.0f / (1.0f + expf(-v));
}

__global__ void scale_to_h(const float* __restrict__ out_pre, const float* __restrict__ scale,
                           _Float16* __restrict__ out_h) {
  int id = blockIdx.x * blockDim.x + threadIdx.x;
  if (id >= NPIX * CC) return;
  out_h[id] = (_Float16)(out_pre[id] * scale[id & (CC - 1)]);
}

// ---------------------------------------------------------------------------
extern "C" void kernel_launch(void* const* d_in, const int* in_sizes, int n_in,
                              void* d_out, int out_size, void* d_ws, size_t ws_size,
                              hipStream_t stream) {
  const float* x         = (const float*)d_in[0];
  const float* raw_sigma = (const float*)d_in[1];
  const float* base_off  = (const float*)d_in[2];
  const float* w_in      = (const float*)d_in[3];
  const float* b_in      = (const float*)d_in[4];
  const float* w_out     = (const float*)d_in[5];
  const float* b_out     = (const float*)d_in[6];
  const float* dw_k      = (const float*)d_in[7];
  const float* dw_b      = (const float*)d_in[8];
  const float* pw_w      = (const float*)d_in[9];
  const float* pw_b      = (const float*)d_in[10];
  const float* off_w     = (const float*)d_in[11];
  const float* off_b     = (const float*)d_in[12];
  const float* msk_w     = (const float*)d_in[13];
  const float* msk_b     = (const float*)d_in[14];
  const float* k1_w      = (const float*)d_in[15];
  const float* k1_b      = (const float*)d_in[16];
  const float* k2_w      = (const float*)d_in[17];
  const float* k2_b      = (const float*)d_in[18];
  const float* k3_w      = (const float*)d_in[19];
  const float* k3_b      = (const float*)d_in[20];
  const float* se1_w     = (const float*)d_in[21];
  const float* se1_b     = (const float*)d_in[22];
  const float* se2_w     = (const float*)d_in[23];
  const float* se2_b     = (const float*)d_in[24];
  float* out = (float*)d_out;

  // ---- workspace layout (256B aligned slices) ----
  char* ws = (char*)d_ws;
  size_t off = 0;
  auto alloc = [&](size_t bytes) -> char* {
    char* p = ws + off;
    off = (off + bytes + 255) & ~(size_t)255;
    return p;
  };
  float*    env      = (float*)alloc(K2 * 4);
  float*    kwbuf    = (float*)alloc(K2 * CC * 4);
  float*    x_proj   = (float*)alloc((size_t)NPIX * CC * 4);
  _Float16* x_h      = (_Float16*)alloc((size_t)NPIX * CC * 2);
  _Float16* y_h      = (_Float16*)alloc((size_t)NPIX * CC * 2);
  _Float16* xdw_h    = (_Float16*)alloc((size_t)NPIX * CC * 2);
  float*    offs     = (float*)alloc((size_t)NPIX * NOFF_PAD * 4);
  float*    mask     = (float*)alloc((size_t)NPIX * NMSK_PAD * 4);
  float*    out_pre  = (float*)alloc((size_t)NPIX * CC * 4);
  _Float16* out_h    = (_Float16*)alloc((size_t)NPIX * CC * 2);
  float*    scale    = (float*)alloc(CC * 4);
  _Float16* w_in_p   = (_Float16*)alloc((size_t)CC * CC * 2);
  _Float16* pw_w_p   = (_Float16*)alloc((size_t)CC * CC * 2);
  _Float16* w_out_p  = (_Float16*)alloc((size_t)CC * CC * 2);
  _Float16* off_w_p  = (_Float16*)alloc((size_t)CC * NOFF_PAD * 2);
  _Float16* msk_w_p  = (_Float16*)alloc((size_t)CC * NMSK_PAD * 2);
  float*    off_b_p  = (float*)alloc(NOFF_PAD * 4);
  float*    msk_b_p  = (float*)alloc(NMSK_PAD * 4);
  (void)ws_size;

  // 1. tiny precompute
  prep_kernel<<<1, 256, 0, stream>>>(raw_sigma, k1_w, k1_b, k2_w, k2_b, k3_w, k3_b,
                                     env, kwbuf);

  // 2. weight packing into WMMA fragment order (+ zero pad)
  auto packb = [&](const float* s, _Float16* d, int N, int Np) {
    int total = (Np >> 4) * 4 * 32;
    pack_b_f16<<<(total + 255) / 256, 256, 0, stream>>>(s, d, N, Np);
  };
  packb(w_in,  w_in_p,  CC, CC);
  packb(pw_w,  pw_w_p,  CC, CC);
  packb(w_out, w_out_p, CC, CC);
  packb(off_w, off_w_p, GG * K2 * 2, NOFF_PAD);
  packb(msk_w, msk_w_p, GG * K2,     NMSK_PAD);
  pad_bias_f32<<<(NOFF_PAD + 255) / 256, 256, 0, stream>>>(off_b, off_b_p,
                                                           GG * K2 * 2, NOFF_PAD);
  pad_bias_f32<<<(NMSK_PAD + 255) / 256, 256, 0, stream>>>(msk_b, msk_b_p,
                                                           GG * K2, NMSK_PAD);

  // 3. depthwise conv + SiLU, x->f16
  dwconv_silu<<<(NPIX * CC) / 256, 256, 0, stream>>>(x, dw_k, dw_b, x_h, y_h);

  // 4. GEMMs (wave32 WMMA, 16x16 tile per wave, 4 waves/block, grid.y = M/16)
  dim3 blk(128);
  auto gemm = [&](const _Float16* A, const _Float16* Bp, const float* bias,
                  const float* scl, float* oF, _Float16* oH, int Npad) {
    int Ntiles = Npad >> 4;
    dim3 grd((Ntiles + 3) / 4, NPIX / 16);
    wmma_gemm<<<grd, blk, 0, stream>>>(A, Bp, bias, scl, oF, oH, Npad);
  };
  gemm(x_h,   w_in_p,  b_in,    nullptr,  x_proj, nullptr, CC);        // x_proj
  gemm(y_h,   pw_w_p,  pw_b,    nullptr,  nullptr, xdw_h,  CC);        // x_dw (f16)
  gemm(xdw_h, off_w_p, off_b_p, base_off, offs,   nullptr, NOFF_PAD);  // offsets*scale
  gemm(xdw_h, msk_w_p, msk_b_p, nullptr,  mask,   nullptr, NMSK_PAD);  // raw mask

  // 5. softmax over taps (4096 waves)
  softmax_mask<<<(NPIX * GG * 32) / 256, 256, 0, stream>>>(mask, env);

  // 6. deformable gather (4096 waves, lane == group-channel)
  deform_sample<<<(NPIX * GG * 32) / 128, 128, 0, stream>>>(x_proj, offs, mask,
                                                            kwbuf, out_pre);

  // 7. SE gate + apply + f16 convert
  se_kernel<<<1, 128, 0, stream>>>(out_pre, se1_w, se1_b, se2_w, se2_b, scale);
  scale_to_h<<<(NPIX * CC) / 256, 256, 0, stream>>>(out_pre, scale, out_h);

  // 8. final projection -> d_out (f32)
  gemm(out_h, w_out_p, b_out, nullptr, out, nullptr, CC);
}
